// Model_82566451298546
// MI455X (gfx1250) — matrix operand order repaired and checked by reading, not compiled, not run on hardware
//
#include <hip/hip_runtime.h>
#include <math.h>

// ---------------------------------------------------------------------------
// CDNA5 / gfx1250 fused attention-variant pipeline.
// All heavy GEMMs run on v_wmma_f32_16x16x32_bf16 (wave32 WMMA).
// Tile staging uses GLOBAL_LOAD_ASYNC_TO_LDS_B128 when the builtin exists.
// ---------------------------------------------------------------------------

typedef __attribute__((ext_vector_type(16))) __bf16       v16bf;
typedef __attribute__((ext_vector_type(8)))  float        v8f;
typedef __attribute__((ext_vector_type(4)))  unsigned int v4u;   // POD 16B chunk
typedef __attribute__((ext_vector_type(4)))  int          v4i;

static constexpr int BB = 4, LQ = 2048, LKK = 2048, DD = 2048;

static constexpr int BLK_M = 128;
static constexpr int BLK_N = 128;
static constexpr int BLK_K = 32;
static constexpr int APITCH    = BLK_K + 16;  // 48 ushorts -> 96B rows (16B aligned)
static constexpr int BPITCH_NT = BLK_K + 16;  // 48
static constexpr int BPITCH_NN = BLK_N + 16;  // 144 ushorts -> 288B rows

#define DEVFN static __device__ __forceinline__

// ---- gfx1250 async global->LDS path (guarded; falls back to ld/st) --------
#if defined(__has_builtin)
#if __has_builtin(__builtin_amdgcn_global_load_async_to_lds_b128)
#define USE_ASYNC_LDS 1
#endif
#endif
#ifndef USE_ASYNC_LDS
#define USE_ASYNC_LDS 0
#endif

#if USE_ASYNC_LDS
// Builtin prototype (from hipcc diagnostic): global <4 x i32>* (AS1),
// LDS <4 x i32>* (AS3), imm offset, imm cpol.
typedef __attribute__((address_space(1))) v4i AS1v4i;
typedef __attribute__((address_space(3))) v4i AS3v4i;

DEVFN void async_cp16(unsigned short* lds, const unsigned short* gp) {
  __builtin_amdgcn_global_load_async_to_lds_b128(
      (AS1v4i*)(gp), (AS3v4i*)(lds), /*offset=*/0, /*cpol=*/0);
}
DEVFN void async_wait0() {
#if __has_builtin(__builtin_amdgcn_s_wait_asynccnt)
  __builtin_amdgcn_s_wait_asynccnt(0);
#else
  asm volatile("s_wait_asynccnt 0x0" ::: "memory");
#endif
}
#endif

DEVFN unsigned short f2bf(float f) {
  unsigned int u = __float_as_uint(f);
  unsigned int r = (u >> 16) & 1u;              // round-to-nearest-even
  return (unsigned short)((u + 0x7FFFu + r) >> 16);
}

// ---- WMMA fragment loaders (contiguous K order shared by A and B, wave32) -------------
// A (16x32 bf16): lane l -> row M=l&15; lanes 0-15 hold K=0..15,
//                 lanes 16-31 hold K=16..31 (contiguous run of 16).
DEVFN v16bf load_a_frag(const unsigned short* s, int pitch, int row0, int lane) {
  const int r = row0 + (lane & 15);
  const int g = (lane >> 4) << 4;                  // 0 or 16
  const unsigned short* p = s + r * pitch + g;
  union { v4u u[2]; v16bf v; } c;
  c.u[0] = *(const v4u*)(p);                        // K = g .. g+7
  c.u[1] = *(const v4u*)(p + 8);                    // K = g+8 .. g+15
  return c.v;
}

// B (32x16 bf16, conceptually K x N): lane l -> col N=l&15;
// lanes 0-15 hold K=0..15, lanes 16-31 hold K=16..31 (contiguous run of 16).
// Bs stored as [N][K] (NT path: rows of the [N,K] operand staged directly).
DEVFN v16bf load_b_frag_nt(const unsigned short* s, int pitch, int col0, int lane) {
  const int n = col0 + (lane & 15);
  const int g = (lane >> 4) << 4;                  // 0 or 16
  const unsigned short* p = s + n * pitch + g;
  union { v4u u[2]; v16bf v; } c;
  c.u[0] = *(const v4u*)(p);                        // K = g .. g+7
  c.u[1] = *(const v4u*)(p + 8);                    // K = g+8 .. g+15
  return c.v;
}

// Bs stored as [K][N] (NN path): gather a column.
DEVFN v16bf load_b_frag_nn(const unsigned short* s, int pitch, int col0, int lane) {
  const int n = col0 + (lane & 15);
  const int g = (lane >> 4) << 4;
  union { unsigned short e[16]; v16bf v; } c;
#pragma unroll
  for (int j = 0; j < 16; ++j) c.e[j] = s[(g + j) * pitch + n];
  return c.v;
}

// ---------------------------------------------------------------------------
// f32 -> bf16 conversion (grid-stride, streaming NT stores)
// ---------------------------------------------------------------------------
__global__ __launch_bounds__(256) void f32_to_bf16_kernel(
    const float* __restrict__ in, unsigned short* __restrict__ out, int n) {
  int i = blockIdx.x * blockDim.x + threadIdx.x;
  const int stride = gridDim.x * blockDim.x;
  for (; i < n; i += stride) {
    unsigned short v = f2bf(in[i]);
    __builtin_nontemporal_store(v, &out[i]);
  }
}

// ---------------------------------------------------------------------------
// NT GEMM:  Out[M,N] = A[M,K] @ W[N,K]^T  (+bias)        -> bf16 out   (proj)
// ---------------------------------------------------------------------------
__global__ __launch_bounds__(256) void proj_gemm_kernel(
    const unsigned short* __restrict__ A,    // [M,K] bf16
    const unsigned short* __restrict__ W,    // [N,K] bf16
    const float* __restrict__ bias,          // [N]
    unsigned short* __restrict__ Out,        // [M,N] bf16
    int M, int N, int K) {
  __shared__ unsigned short As[BLK_M * APITCH];
  __shared__ unsigned short Bs[BLK_N * BPITCH_NT];

  const int tid = threadIdx.x, lane = tid & 31, wave = tid >> 5;
  const int wm = (wave >> 2) * 64;   // 0 / 64
  const int wn = (wave & 3) * 32;    // 0,32,64,96
  const int m0 = blockIdx.y * BLK_M;
  const int n0 = blockIdx.x * BLK_N;

  v8f zero = {};
  v8f acc[4][2];
#pragma unroll
  for (int i = 0; i < 4; ++i)
#pragma unroll
    for (int j = 0; j < 2; ++j) acc[i][j] = zero;

  const int lr = tid >> 1;                 // 0..127
  const int lc = (tid & 1) << 4;           // 0 / 16
  const int ktiles = K / BLK_K;

  for (int kt = 0; kt < ktiles; ++kt) {
    const int k0 = kt * BLK_K;
    const unsigned short* ga = A + (size_t)(m0 + lr) * K + k0 + lc;
    const unsigned short* gb = W + (size_t)(n0 + lr) * K + k0 + lc;
#if USE_ASYNC_LDS
    async_cp16(&As[lr * APITCH + lc],         ga);
    async_cp16(&As[lr * APITCH + lc + 8],     ga + 8);
    async_cp16(&Bs[lr * BPITCH_NT + lc],      gb);
    async_cp16(&Bs[lr * BPITCH_NT + lc + 8],  gb + 8);
#else
    *(v4u*)(&As[lr * APITCH + lc])        = *(const v4u*)(ga);
    *(v4u*)(&As[lr * APITCH + lc + 8])    = *(const v4u*)(ga + 8);
    *(v4u*)(&Bs[lr * BPITCH_NT + lc])     = *(const v4u*)(gb);
    *(v4u*)(&Bs[lr * BPITCH_NT + lc + 8]) = *(const v4u*)(gb + 8);
#endif
    if (kt + 1 < ktiles) {  // gfx1250 global_prefetch_b8 for next K tile
      __builtin_prefetch(ga + BLK_K, 0, 0);
      __builtin_prefetch(gb + BLK_K, 0, 0);
    }
#if USE_ASYNC_LDS
    async_wait0();
#endif
    __syncthreads();

    v16bf af[4], bfg[2];
#pragma unroll
    for (int i = 0; i < 4; ++i) af[i] = load_a_frag(As, APITCH, wm + i * 16, lane);
#pragma unroll
    for (int j = 0; j < 2; ++j) bfg[j] = load_b_frag_nt(Bs, BPITCH_NT, wn + j * 16, lane);
#pragma unroll
    for (int i = 0; i < 4; ++i)
#pragma unroll
      for (int j = 0; j < 2; ++j)
        acc[i][j] = __builtin_amdgcn_wmma_f32_16x16x32_bf16(
            false, af[i], false, bfg[j], (short)0, acc[i][j], false, false);
    __syncthreads();
  }

  const int crow = wm + ((lane < 16) ? 0 : 8);
  const int ccol = wn + (lane & 15);
#pragma unroll
  for (int i = 0; i < 4; ++i)
#pragma unroll
    for (int j = 0; j < 2; ++j) {
      const int gm = m0 + crow + i * 16;
      const int gn = n0 + ccol + j * 16;
      const float bv = bias[gn];
#pragma unroll
      for (int r = 0; r < 8; ++r)
        Out[(size_t)(gm + r) * N + gn] = f2bf(acc[i][j][r] + bv);
    }
}

// ---------------------------------------------------------------------------
// NT GEMM per batch:  Sc[b,q,k] = (Q[b] @ K[b]^T) * scale + mask[q,k]  -> f32
// ---------------------------------------------------------------------------
__global__ __launch_bounds__(256) void scores_gemm_kernel(
    const unsigned short* __restrict__ Q,    // [B*LQ, D] bf16
    const unsigned short* __restrict__ Kb,   // [B*LK, D] bf16
    const float* __restrict__ mask,          // [LQ, LK] f32
    float* __restrict__ Sc,                  // [B*LQ, LK] f32
    float scale) {
  __shared__ unsigned short As[BLK_M * APITCH];
  __shared__ unsigned short Bs[BLK_N * BPITCH_NT];

  const int tid = threadIdx.x, lane = tid & 31, wave = tid >> 5;
  const int wm = (wave >> 2) * 64;
  const int wn = (wave & 3) * 32;
  const int m0 = blockIdx.y * BLK_M;
  const int n0 = blockIdx.x * BLK_N;
  const int bz = blockIdx.z;
  const unsigned short* A = Q  + (size_t)bz * LQ  * DD;
  const unsigned short* W = Kb + (size_t)bz * LKK * DD;

  v8f zero = {};
  v8f acc[4][2];
#pragma unroll
  for (int i = 0; i < 4; ++i)
#pragma unroll
    for (int j = 0; j < 2; ++j) acc[i][j] = zero;

  const int lr = tid >> 1;
  const int lc = (tid & 1) << 4;
  const int ktiles = DD / BLK_K;

  for (int kt = 0; kt < ktiles; ++kt) {
    const int k0 = kt * BLK_K;
    const unsigned short* ga = A + (size_t)(m0 + lr) * DD + k0 + lc;
    const unsigned short* gb = W + (size_t)(n0 + lr) * DD + k0 + lc;
#if USE_ASYNC_LDS
    async_cp16(&As[lr * APITCH + lc],         ga);
    async_cp16(&As[lr * APITCH + lc + 8],     ga + 8);
    async_cp16(&Bs[lr * BPITCH_NT + lc],      gb);
    async_cp16(&Bs[lr * BPITCH_NT + lc + 8],  gb + 8);
#else
    *(v4u*)(&As[lr * APITCH + lc])        = *(const v4u*)(ga);
    *(v4u*)(&As[lr * APITCH + lc + 8])    = *(const v4u*)(ga + 8);
    *(v4u*)(&Bs[lr * BPITCH_NT + lc])     = *(const v4u*)(gb);
    *(v4u*)(&Bs[lr * BPITCH_NT + lc + 8]) = *(const v4u*)(gb + 8);
#endif
    if (kt + 1 < ktiles) {
      __builtin_prefetch(ga + BLK_K, 0, 0);
      __builtin_prefetch(gb + BLK_K, 0, 0);
    }
#if USE_ASYNC_LDS
    async_wait0();
#endif
    __syncthreads();

    v16bf af[4], bfg[2];
#pragma unroll
    for (int i = 0; i < 4; ++i) af[i] = load_a_frag(As, APITCH, wm + i * 16, lane);
#pragma unroll
    for (int j = 0; j < 2; ++j) bfg[j] = load_b_frag_nt(Bs, BPITCH_NT, wn + j * 16, lane);
#pragma unroll
    for (int i = 0; i < 4; ++i)
#pragma unroll
      for (int j = 0; j < 2; ++j)
        acc[i][j] = __builtin_amdgcn_wmma_f32_16x16x32_bf16(
            false, af[i], false, bfg[j], (short)0, acc[i][j], false, false);
    __syncthreads();
  }

  const int crow = wm + ((lane < 16) ? 0 : 8);
  const int ccol = wn + (lane & 15);
#pragma unroll
  for (int i = 0; i < 4; ++i)
#pragma unroll
    for (int j = 0; j < 2; ++j) {
      const int gq = m0 + crow + i * 16;
      const int gk = n0 + ccol + j * 16;
#pragma unroll
      for (int r = 0; r < 8; ++r) {
        float v = acc[i][j][r] * scale + mask[(size_t)(gq + r) * LKK + gk];
        // scores are stream-once data (64 MB): NT store keeps Q/K tiles in L2
        __builtin_nontemporal_store(v, &Sc[((size_t)bz * LQ + gq + r) * LKK + gk]);
      }
    }
}

// ---------------------------------------------------------------------------
// NN GEMM per batch:  Out[b,q,d] = mask[q,:] @ V[b,:,d]   -> f32 (into d_out)
// ---------------------------------------------------------------------------
__global__ __launch_bounds__(256) void mixed_gemm_kernel(
    const unsigned short* __restrict__ Mb,   // [LQ, LK] bf16 mask
    const unsigned short* __restrict__ Vb,   // [B*LK, D] bf16
    float* __restrict__ Out) {               // [B*LQ, D] f32
  __shared__ unsigned short As[BLK_M * APITCH];
  __shared__ unsigned short Bs[BLK_K * BPITCH_NN];

  const int tid = threadIdx.x, lane = tid & 31, wave = tid >> 5;
  const int wm = (wave >> 2) * 64;
  const int wn = (wave & 3) * 32;
  const int m0 = blockIdx.y * BLK_M;
  const int n0 = blockIdx.x * BLK_N;
  const int bz = blockIdx.z;
  const unsigned short* V = Vb + (size_t)bz * LKK * DD;

  v8f zero = {};
  v8f acc[4][2];
#pragma unroll
  for (int i = 0; i < 4; ++i)
#pragma unroll
    for (int j = 0; j < 2; ++j) acc[i][j] = zero;

  const int ar = tid >> 1, ac = (tid & 1) << 4;   // A tile: 128x32
  const int br = tid >> 3, bc = (tid & 7) << 4;   // B tile: 32x128
  const int ktiles = LKK / BLK_K;

  for (int kt = 0; kt < ktiles; ++kt) {
    const int k0 = kt * BLK_K;
    const unsigned short* ga = Mb + (size_t)(m0 + ar) * LKK + k0 + ac;
    const unsigned short* gb = V + (size_t)(k0 + br) * DD + n0 + bc;
#if USE_ASYNC_LDS
    async_cp16(&As[ar * APITCH + ac],        ga);
    async_cp16(&As[ar * APITCH + ac + 8],    ga + 8);
    async_cp16(&Bs[br * BPITCH_NN + bc],     gb);
    async_cp16(&Bs[br * BPITCH_NN + bc + 8], gb + 8);
#else
    *(v4u*)(&As[ar * APITCH + ac])        = *(const v4u*)(ga);
    *(v4u*)(&As[ar * APITCH + ac + 8])    = *(const v4u*)(ga + 8);
    *(v4u*)(&Bs[br * BPITCH_NN + bc])     = *(const v4u*)(gb);
    *(v4u*)(&Bs[br * BPITCH_NN + bc + 8]) = *(const v4u*)(gb + 8);
#endif
    if (kt + 1 < ktiles) {
      __builtin_prefetch(ga + BLK_K, 0, 0);
      __builtin_prefetch(V + (size_t)(k0 + BLK_K + br) * DD + n0 + bc, 0, 0);
    }
#if USE_ASYNC_LDS
    async_wait0();
#endif
    __syncthreads();

    v16bf af[4], bfg[2];
#pragma unroll
    for (int i = 0; i < 4; ++i) af[i] = load_a_frag(As, APITCH, wm + i * 16, lane);
#pragma unroll
    for (int j = 0; j < 2; ++j) bfg[j] = load_b_frag_nn(Bs, BPITCH_NN, wn + j * 16, lane);
#pragma unroll
    for (int i = 0; i < 4; ++i)
#pragma unroll
      for (int j = 0; j < 2; ++j)
        acc[i][j] = __builtin_amdgcn_wmma_f32_16x16x32_bf16(
            false, af[i], false, bfg[j], (short)0, acc[i][j], false, false);
    __syncthreads();
  }

  const int crow = wm + ((lane < 16) ? 0 : 8);
  const int ccol = wn + (lane & 15);
#pragma unroll
  for (int i = 0; i < 4; ++i)
#pragma unroll
    for (int j = 0; j < 2; ++j) {
      const int gq = m0 + crow + i * 16;
      const int gd = n0 + ccol + j * 16;
#pragma unroll
      for (int r = 0; r < 8; ++r)
        Out[((size_t)bz * LQ + gq + r) * DD + gd] = acc[i][j][r];
    }
}

// ---------------------------------------------------------------------------
// Row softmax over scores + elementwise combine into d_out (in place).
// One 256-thread (8-wave) block per (b,q) row; wave32 shuffle reductions.
// ---------------------------------------------------------------------------
__global__ __launch_bounds__(256) void softmax_combine_kernel(
    const float* __restrict__ Sc,   // [B*LQ, LK]
    float* __restrict__ Out,        // [B*LQ, D], D == LK
    int lk) {
  const int row = blockIdx.x;
  const float* s = Sc + (size_t)row * lk;
  float* o = Out + (size_t)row * lk;
  __shared__ float red[8];
  const int tid = threadIdx.x, lane = tid & 31, wave = tid >> 5;

  float m = -3.402823466e38f;
  for (int i = tid; i < lk; i += 256) m = fmaxf(m, s[i]);
#pragma unroll
  for (int off = 16; off; off >>= 1) m = fmaxf(m, __shfl_xor(m, off, 32));
  if (lane == 0) red[wave] = m;
  __syncthreads();
  m = red[0];
#pragma unroll
  for (int w = 1; w < 8; ++w) m = fmaxf(m, red[w]);
  __syncthreads();

  float sum = 0.f;
  for (int i = tid; i < lk; i += 256) sum += __expf(s[i] - m);
#pragma unroll
  for (int off = 16; off; off >>= 1) sum += __shfl_xor(sum, off, 32);
  if (lane == 0) red[wave] = sum;
  __syncthreads();
  sum = 0.f;
#pragma unroll
  for (int w = 0; w < 8; ++w) sum += red[w];
  const float inv = 1.f / sum;

  for (int i = tid; i < lk; i += 256) o[i] = o[i] * (__expf(s[i] - m) * inv);
}

// ---------------------------------------------------------------------------
// Host-side launcher
// ---------------------------------------------------------------------------
extern "C" void kernel_launch(void* const* d_in, const int* in_sizes, int n_in,
                              void* d_out, int out_size, void* d_ws, size_t ws_size,
                              hipStream_t stream) {
  (void)in_sizes; (void)n_in; (void)out_size; (void)ws_size;
  const float* query = (const float*)d_in[0];
  const float* key   = (const float*)d_in[1];
  const float* value = (const float*)d_in[2];
  const float* Wq    = (const float*)d_in[3];
  const float* bq    = (const float*)d_in[4];
  const float* Wk    = (const float*)d_in[5];
  const float* bk    = (const float*)d_in[6];
  const float* mask  = (const float*)d_in[7];
  float* out = (float*)d_out;

  char* ws = (char*)d_ws;
  size_t off = 0;
  auto carve = [&](size_t bytes) -> void* {
    void* p = ws + off;
    off += (bytes + 255) & ~(size_t)255;
    return p;
  };
  const size_t nBQ = (size_t)BB * LQ * DD;      // 16.7M elements
  const size_t nW  = (size_t)DD * DD;           // 4.2M
  unsigned short* Xq  = (unsigned short*)carve(nBQ * 2);
  unsigned short* Xk  = (unsigned short*)carve(nBQ * 2);
  unsigned short* Vb  = (unsigned short*)carve(nBQ * 2);
  unsigned short* Wqb = (unsigned short*)carve(nW * 2);
  unsigned short* Wkb = (unsigned short*)carve(nW * 2);
  unsigned short* Mb  = (unsigned short*)carve((size_t)LQ * LKK * 2);
  unsigned short* Qb  = (unsigned short*)carve(nBQ * 2);
  unsigned short* Kb  = (unsigned short*)carve(nBQ * 2);
  float*          Sc  = (float*)carve((size_t)BB * LQ * LKK * 4);

  // 1) f32 -> bf16 conversions
  f32_to_bf16_kernel<<<2048, 256, 0, stream>>>(query, Xq,  (int)nBQ);
  f32_to_bf16_kernel<<<2048, 256, 0, stream>>>(key,   Xk,  (int)nBQ);
  f32_to_bf16_kernel<<<2048, 256, 0, stream>>>(value, Vb,  (int)nBQ);
  f32_to_bf16_kernel<<<1024, 256, 0, stream>>>(Wq,    Wqb, (int)nW);
  f32_to_bf16_kernel<<<1024, 256, 0, stream>>>(Wk,    Wkb, (int)nW);
  f32_to_bf16_kernel<<<1024, 256, 0, stream>>>(mask,  Mb,  (int)((size_t)LQ * LKK));

  // 2) projections: Q = Xq@Wq^T+bq, K = Xk@Wk^T+bk  (M=8192, N=K=2048)
  dim3 pg(DD / BLK_N, (BB * LQ) / BLK_M, 1);
  proj_gemm_kernel<<<pg, 256, 0, stream>>>(Xq, Wqb, bq, Qb, BB * LQ, DD, DD);
  proj_gemm_kernel<<<pg, 256, 0, stream>>>(Xk, Wkb, bk, Kb, BB * LKK, DD, DD);

  // 3) scores = Q@K^T * scale + mask  (per batch)
  const float scale = 1.0f / sqrtf((float)DD);
  dim3 sg(LKK / BLK_N, LQ / BLK_M, BB);
  scores_gemm_kernel<<<sg, 256, 0, stream>>>(Qb, Kb, mask, Sc, scale);

  // 4) mixed = mask @ V  (per batch) -> d_out
  dim3 mg(DD / BLK_N, LQ / BLK_M, BB);
  mixed_gemm_kernel<<<mg, 256, 0, stream>>>(Mb, Vb, out);

  // 5) out = mixed * softmax(scores)
  softmax_combine_kernel<<<BB * LQ, 256, 0, stream>>>(Sc, out, LKK);
}
